// FullAttention_71399536329293
// MI455X (gfx1250) — matrix operand order repaired and checked by reading, not compiled, not run on hardware
//
#include <hip/hip_runtime.h>
#include <cmath>

// ---------------------------------------------------------------- types
typedef unsigned short ushortT;
typedef __attribute__((ext_vector_type(8)))  float          v8f;
typedef __attribute__((ext_vector_type(8)))  unsigned short v8u;
typedef __attribute__((ext_vector_type(16))) unsigned short v16u;
typedef __attribute__((ext_vector_type(16))) __bf16         v16bf;
typedef __attribute__((ext_vector_type(4)))  unsigned int   v4u32;
typedef __attribute__((ext_vector_type(8)))  int            v8i32;
typedef __attribute__((ext_vector_type(4)))  int            v4i32;

#define HID      768
#define HEADS    12
#define HEAD_DIM 64
#define MLP      3072
#define FUSED_N  5376   // MLP + 3*HID
#define S_TOT    2048   // 16*16*8 tokens

// TDM dispatch (probe-verified arities):
//   amdgpu-toolchain (ships amd_gfx1250_TDM.h): 6-arg builtin
//   ROCm 7.2 (no header):                       5-arg builtin
//   builtin absent (incl. host pass):           manual global->LDS fallback
#if __has_builtin(__builtin_amdgcn_tensor_load_to_lds)
#define USE_TDM 1
#if __has_include(<hip/amd_detail/amd_gfx1250_TDM.h>)
#define TDM_SIX 1
#else
#define TDM_SIX 0
#endif
#else
#define USE_TDM 0
#endif

// ---------------------------------------------------------------- helpers
__device__ __forceinline__ ushortT f2bf(float f) {
  unsigned u = __builtin_bit_cast(unsigned, f);
  u += 0x7FFFu + ((u >> 16) & 1u);          // round-to-nearest-even
  return (ushortT)(u >> 16);
}

__device__ __forceinline__ v8u ld8(const ushortT* p) { return *(const v8u*)p; }

__device__ __forceinline__ v16u cat16(v8u lo, v8u hi) {
  return __builtin_shufflevector(lo, hi, 0, 1, 2, 3, 4, 5, 6, 7,
                                 8, 9, 10, 11, 12, 13, 14, 15);
}
__device__ __forceinline__ v16u ld16(const ushortT* p) {
  return cat16(*(const v8u*)p, *(const v8u*)(p + 8));
}

// D = A(16x32 bf16) * B(32x16 bf16) + C(16x16 f32)
__device__ __forceinline__ v8f wmma_bf16(v16u a, v16u b, v8f c) {
  return __builtin_amdgcn_wmma_f32_16x16x32_bf16(
      false, __builtin_bit_cast(v16bf, a),
      false, __builtin_bit_cast(v16bf, b),
      (short)0, c, false, false);
}

__device__ __forceinline__ float wave_red_sum(float v) {
#pragma unroll
  for (int m = 1; m <= 16; m <<= 1) v += __shfl_xor(v, m, 32);
  return v;
}

#if USE_TDM
// Build a D# (ISA §8) for a 2D bf16 tile: 32 (x, contiguous K) x 128 (y, rows),
// row stride = stride0 elements, tensor bounds d0 x d1, and fire the TDM.
__device__ __forceinline__ void tdm_load_tile(unsigned lds_byte,
                                              const ushortT* gtile,
                                              unsigned d0, unsigned d1,
                                              unsigned stride0) {
  unsigned long long ga = (unsigned long long)(size_t)gtile;
  v4u32 g0;
  g0.x = 1u;                                              // count=1, user, no gather
  g0.y = lds_byte;                                        // lds_addr
  g0.z = (unsigned)(ga & 0xffffffffu);                    // global_addr[31:0]
  g0.w = (unsigned)((ga >> 32) & 0x01ffffffu) | (2u << 30); // addr[56:32] | type=2
  v8i32 g1;
  g1[0] = (int)(1u << 16);                                // data_size=1 -> 2 bytes
  g1[1] = (int)((d0 & 0xffffu) << 16);                    // tensor_dim0[15:0]
  g1[2] = (int)((d0 >> 16) | ((d1 & 0xffffu) << 16));     // dim0 hi | dim1 lo
  g1[3] = (int)((d1 >> 16) | (32u << 16));                // dim1 hi | tile_dim0=32
  g1[4] = (int)(128u);                                    // tile_dim1=128, tile_dim2=0
  g1[5] = (int)stride0;                                   // tensor_dim0_stride[31:0]
  g1[6] = 0;                                              // stride0 hi | stride1 lo
  g1[7] = 0;
  v4i32 z4 = {0, 0, 0, 0};
#if TDM_SIX
  v8i32 z8 = {0, 0, 0, 0, 0, 0, 0, 0};
  __builtin_amdgcn_tensor_load_to_lds(g0, g1, z4, z4, z8, 0);
#else
  __builtin_amdgcn_tensor_load_to_lds(g0, g1, z4, z4, 0);
#endif
}
#endif

// ---------------------------------------------------------------- f32 -> bf16, transposed
// dst[N][K] bf16 = transpose(src[K][N] f32)
__global__ __launch_bounds__(256) void cvt_bf16_t(const float* __restrict__ src,
                                                  ushortT* __restrict__ dst,
                                                  int K, int N) {
  int i = blockIdx.x * blockDim.x + threadIdx.x;
  int stride = gridDim.x * blockDim.x;
  const int total = K * N;
  for (; i < total; i += stride) {
    int n = i / K, k = i - n * K;
    dst[i] = f2bf(src[(size_t)k * N + n]);
  }
}

// ---------------------------------------------------------------- group RMS norm + transpose
// x: [768][2048] f32 -> xn: [2048][768] bf16, 12 groups of 64 channels
__global__ __launch_bounds__(256) void gnorm_kernel(const float* __restrict__ x,
                                                    const float* __restrict__ gamma,
                                                    ushortT* __restrict__ xn) {
  const int lane = threadIdx.x & 31, w = threadIdx.x >> 5;
  const int gid = blockIdx.x * 8 + w;
  const int s = gid / 12, g = gid % 12;
  const int c = g * 64 + lane * 2;
  float a = x[(size_t)c * S_TOT + s];
  float b = x[(size_t)(c + 1) * S_TOT + s];
  float ss = wave_red_sum(a * a + b * b);
  float r = rsqrtf(ss * (1.0f / 64.0f) + 1e-6f);
  xn[(size_t)s * HID + c]     = f2bf(a * r * gamma[c]);
  xn[(size_t)s * HID + c + 1] = f2bf(b * r * gamma[c + 1]);
}

// ---------------------------------------------------------------- WMMA GEMM
// C[M][N] = A[M][K](bf16) * Bt[N][K](bf16, pre-transposed) (+ bias[N])
// block tile 128x128, wave tile 32x64, K-step 32, TDM double-buffered
__global__ __launch_bounds__(256) void gemm_bf16(const ushortT* __restrict__ A,
                                                 const ushortT* __restrict__ Bt,
                                                 float* __restrict__ C,
                                                 const float* __restrict__ bias,
                                                 int M, int N, int K) {
  __shared__ __align__(16) ushortT sA[2][128 * 32];
  __shared__ __align__(16) ushortT sB[2][128 * 32];
  const int tid = threadIdx.x;
  const int lane = tid & 31, w = tid >> 5;
  const int lrow = lane & 15, hh = lane >> 4;      // half-wave id
  const int wm = w >> 1, wn = w & 1;               // 4x2 wave grid
  const int m0 = blockIdx.y * 128, n0 = blockIdx.x * 128;
  const int nk = K >> 5;

  v8f acc[2][4];
  const v8f vz = {0.f, 0.f, 0.f, 0.f, 0.f, 0.f, 0.f, 0.f};
#pragma unroll
  for (int mt = 0; mt < 2; ++mt)
#pragma unroll
    for (int nt = 0; nt < 4; ++nt) acc[mt][nt] = vz;

#if USE_TDM
  const unsigned la[2] = {(unsigned)(size_t)&sA[0][0], (unsigned)(size_t)&sA[1][0]};
  const unsigned lb[2] = {(unsigned)(size_t)&sB[0][0], (unsigned)(size_t)&sB[1][0]};
  if (tid < 32) {                                   // wave 0 drives the TDM
    tdm_load_tile(la[0], A + (size_t)m0 * K, (unsigned)K, (unsigned)M, (unsigned)K);
    tdm_load_tile(lb[0], Bt + (size_t)n0 * K, (unsigned)K, (unsigned)N, (unsigned)K);
  }
#else
  const int ldrow = tid >> 1, ldseg = (tid & 1) * 16;
#endif

  for (int it = 0; it < nk; ++it) {
    const int cur = it & 1;
#if USE_TDM
    if (tid < 32) {
      if (it + 1 < nk) {                            // prefetch next pair via TDM
        tdm_load_tile(la[cur ^ 1], A + (size_t)m0 * K + (it + 1) * 32,
                      (unsigned)K, (unsigned)M, (unsigned)K);
        tdm_load_tile(lb[cur ^ 1], Bt + (size_t)n0 * K + (it + 1) * 32,
                      (unsigned)K, (unsigned)N, (unsigned)K);
        __builtin_amdgcn_s_wait_tensorcnt(2);       // current pair landed (in-order)
      } else {
        __builtin_amdgcn_s_wait_tensorcnt(0);
      }
    }
#else
    {
      const int k0 = it * 32;
      const ushortT* ag = A + (size_t)(m0 + ldrow) * K + k0 + ldseg;
      const ushortT* bg = Bt + (size_t)(n0 + ldrow) * K + k0 + ldseg;
      v8u a0 = ld8(ag), a1 = ld8(ag + 8);
      v8u b0 = ld8(bg), b1 = ld8(bg + 8);
      if (it + 1 < nk) {
        __builtin_prefetch(ag + 32, 0, 3);          // global_prefetch next K tile
        __builtin_prefetch(bg + 32, 0, 3);
      }
      *(v8u*)&sA[cur][ldrow * 32 + ldseg]     = a0;
      *(v8u*)&sA[cur][ldrow * 32 + ldseg + 8] = a1;
      *(v8u*)&sB[cur][ldrow * 32 + ldseg]     = b0;
      *(v8u*)&sB[cur][ldrow * 32 + ldseg + 8] = b1;
    }
#endif
    __syncthreads();

    v16u af[2], bf4[4];
#pragma unroll
    for (int mt = 0; mt < 2; ++mt) {                // A frag: lane = M row
      int r = wm * 32 + mt * 16 + lrow;
      af[mt] = cat16(*(const v8u*)&sA[cur][r * 32 + 8 * hh],
                     *(const v8u*)&sA[cur][r * 32 + 16 + 8 * hh]);
    }
#pragma unroll
    for (int nt = 0; nt < 4; ++nt) {                // B frag: lane = N col
      int c = wn * 64 + nt * 16 + lrow;
      bf4[nt] = cat16(ld8(&sB[cur][c * 32 + 8 * hh]), ld8(&sB[cur][c * 32 + 16 + 8 * hh]));
    }
#pragma unroll
    for (int mt = 0; mt < 2; ++mt)
#pragma unroll
      for (int nt = 0; nt < 4; ++nt)
        acc[mt][nt] = wmma_bf16(af[mt], bf4[nt], acc[mt][nt]);
    __syncthreads();
  }

#pragma unroll
  for (int mt = 0; mt < 2; ++mt)
#pragma unroll
    for (int nt = 0; nt < 4; ++nt) {
      int row = m0 + wm * 32 + mt * 16;
      int col = n0 + wn * 64 + nt * 16 + lrow;
      float bv = bias ? bias[col] : 0.0f;
#pragma unroll
      for (int i = 0; i < 8; ++i)
        C[(size_t)(row + i + 8 * hh) * N + col] = acc[mt][nt][i] + bv;
    }
}

// ---------------------------------------------------------------- QK layernorm + RoPE, V transpose
// fused: [2048][5376] f32 -> q,k: [12][2048][64] bf16, vT: [12][64][2048] bf16
__global__ __launch_bounds__(256) void qkv_prep(const float* __restrict__ fused,
                                                const float* __restrict__ qw,
                                                const float* __restrict__ qb,
                                                const float* __restrict__ kw,
                                                const float* __restrict__ kb,
                                                ushortT* __restrict__ qo,
                                                ushortT* __restrict__ ko,
                                                ushortT* __restrict__ vt) {
  const int lane = threadIdx.x & 31, w = threadIdx.x >> 5;
  const int gid = blockIdx.x * 8 + w;
  const int s = gid / HEADS, h = gid % HEADS;
  const float* base = fused + (size_t)s * FUSED_N;
  const int c = lane * 2;

  float q0 = base[MLP + h * 64 + c],           q1 = base[MLP + h * 64 + c + 1];
  float k0 = base[MLP + HID + h * 64 + c],     k1 = base[MLP + HID + h * 64 + c + 1];
  float v0 = base[MLP + 2 * HID + h * 64 + c], v1 = base[MLP + 2 * HID + h * 64 + c + 1];

  float qm = wave_red_sum(q0 + q1) * (1.0f / 64.0f);
  float qd0 = q0 - qm, qd1 = q1 - qm;
  float qv = wave_red_sum(qd0 * qd0 + qd1 * qd1) * (1.0f / 64.0f);
  float qr = rsqrtf(qv + 1e-5f);
  q0 = qd0 * qr * qw[c] + qb[c];
  q1 = qd1 * qr * qw[c + 1] + qb[c + 1];

  float km = wave_red_sum(k0 + k1) * (1.0f / 64.0f);
  float kd0 = k0 - km, kd1 = k1 - km;
  float kv = wave_red_sum(kd0 * kd0 + kd1 * kd1) * (1.0f / 64.0f);
  float kr = rsqrtf(kv + 1e-5f);
  k0 = kd0 * kr * kw[c] + kb[c];
  k1 = kd1 * kr * kw[c + 1] + kb[c + 1];

  const int dd = s & 7, ww = (s >> 3) & 15, hh2 = s >> 7;
  const int j = lane;
  float cs = 1.0f, sn = 0.0f;
  if (j < 24) {
    int axis = j >> 3, jj = j & 7;
    float fr = (1.0f + jj * (127.0f / 7.0f)) * 3.14159265358979323846f;
    float pos = (axis == 0) ? (-1.0f + hh2 * (2.0f / 15.0f))
              : (axis == 1) ? (-1.0f + ww * (2.0f / 15.0f))
                            : (-1.0f + dd * (2.0f / 7.0f));
    float f = pos * fr;
    cs = cosf(f);
    sn = sinf(f);
  }
  float nq0 = q0 * cs - q1 * sn, nq1 = q1 * cs + q0 * sn;
  float nk0 = k0 * cs - k1 * sn, nk1 = k1 * cs + k0 * sn;

  size_t ob = ((size_t)h * S_TOT + s) * 64 + c;
  qo[ob] = f2bf(nq0); qo[ob + 1] = f2bf(nq1);
  ko[ob] = f2bf(nk0); ko[ob + 1] = f2bf(nk1);
  vt[((size_t)h * 64 + c) * S_TOT + s]     = f2bf(v0);
  vt[((size_t)h * 64 + c + 1) * S_TOT + s] = f2bf(v1);
}

// ---------------------------------------------------------------- silu(gate)*xh
__global__ __launch_bounds__(256) void ff_act(const float* __restrict__ fused,
                                              ushortT* __restrict__ out) {
  const int n = S_TOT * (MLP / 2);
  int i = blockIdx.x * blockDim.x + threadIdx.x;
  int stride = gridDim.x * blockDim.x;
  for (; i < n; i += stride) {
    int s = i / (MLP / 2), j = i - s * (MLP / 2);
    float xh = fused[(size_t)s * FUSED_N + j];
    float g  = fused[(size_t)s * FUSED_N + (MLP / 2) + j];
    float si = g / (1.0f + __expf(-g));
    out[i] = f2bf(si * xh);
  }
}

// ---------------------------------------------------------------- flash attention (WMMA)
// q,k: [12][2048][64] bf16, vT: [12][64][2048] bf16 -> o: [2048][768] bf16
__global__ __launch_bounds__(256) void attn_kernel(const ushortT* __restrict__ qg,
                                                   const ushortT* __restrict__ kg,
                                                   const ushortT* __restrict__ vg,
                                                   ushortT* __restrict__ og) {
  __shared__ __align__(16) ushortT ldsp[8 * 16 * 32];  // per-wave 16x32 P bounce
  const int tid = threadIdx.x, lane = tid & 31, w = tid >> 5;
  const int lrow = lane & 15, hh = lane >> 4;
  const int h = blockIdx.x >> 4;
  const int q0 = (blockIdx.x & 15) * 128 + w * 16;
  const ushortT* qh = qg + (size_t)h * S_TOT * 64;
  const ushortT* kh = kg + (size_t)h * S_TOT * 64;
  const ushortT* vh = vg + (size_t)h * 64 * S_TOT;
  ushortT* pl = &ldsp[w * 512];

  v16u qf[2];
#pragma unroll
  for (int kk = 0; kk < 2; ++kk) {
    const ushortT* p = qh + (size_t)(q0 + lrow) * 64 + kk * 32 + 8 * hh;
    qf[kk] = cat16(ld8(p), ld8(p + 16));
  }

  const v8f vz = {0.f, 0.f, 0.f, 0.f, 0.f, 0.f, 0.f, 0.f};
  v8f oacc[4];
#pragma unroll
  for (int n = 0; n < 4; ++n) oacc[n] = vz;
  float mi[8], li[8];
#pragma unroll
  for (int i = 0; i < 8; ++i) { mi[i] = -1e30f; li[i] = 0.0f; }

  for (int kbse = 0; kbse < S_TOT; kbse += 32) {
    if (kbse + 32 < S_TOT) {                     // hint next K/V chunk
      __builtin_prefetch(kh + (size_t)(kbse + 32 + lrow) * 64, 0, 3);
      __builtin_prefetch(vh + (size_t)lrow * S_TOT + kbse + 32, 0, 3);
    }
    v8f s0 = vz, s1 = vz;
#pragma unroll
    for (int kk = 0; kk < 2; ++kk) {
      const ushortT* kp0 = kh + (size_t)(kbse + lrow) * 64 + kk * 32 + 8 * hh;
      const ushortT* kp1 = kh + (size_t)(kbse + 16 + lrow) * 64 + kk * 32 + 8 * hh;
      v16u b0 = cat16(ld8(kp0), ld8(kp0 + 16));
      v16u b1 = cat16(ld8(kp1), ld8(kp1 + 16));
      s0 = wmma_bf16(qf[kk], b0, s0);
      s1 = wmma_bf16(qf[kk], b1, s1);
    }
    float p0[8], p1[8];
#pragma unroll
    for (int i = 0; i < 8; ++i) {
      float a = s0[i] * 0.125f, b = s1[i] * 0.125f;
      p0[i] = a; p1[i] = b;
      float r = fmaxf(a, b);
#pragma unroll
      for (int m = 1; m <= 8; m <<= 1) r = fmaxf(r, __shfl_xor(r, m, 32));
      float mn = fmaxf(mi[i], r);
      float alpha = __expf(mi[i] - mn);
      mi[i] = mn;
      p0[i] = __expf(p0[i] - mn);
      p1[i] = __expf(p1[i] - mn);
      float rs = p0[i] + p1[i];
#pragma unroll
      for (int m = 1; m <= 8; m <<= 1) rs += __shfl_xor(rs, m, 32);
      li[i] = li[i] * alpha + rs;
#pragma unroll
      for (int n = 0; n < 4; ++n) oacc[n][i] *= alpha;
    }
#pragma unroll
    for (int i = 0; i < 8; ++i) {                // C-layout -> A-layout via LDS
      pl[(i + 8 * hh) * 32 + lrow]      = f2bf(p0[i]);
      pl[(i + 8 * hh) * 32 + 16 + lrow] = f2bf(p1[i]);
    }
    __syncthreads();
    const ushortT* pp = pl + lrow * 32 + 8 * hh;
    v16u pf = cat16(*(const v8u*)pp, *(const v8u*)(pp + 16));
#pragma unroll
    for (int n = 0; n < 4; ++n) {
      const ushortT* vp = vh + (size_t)(n * 16 + lrow) * S_TOT + kbse + 8 * hh;
      v16u bv = cat16(ld8(vp), ld8(vp + 16));
      oacc[n] = wmma_bf16(pf, bv, oacc[n]);
    }
    __syncthreads();
  }

#pragma unroll
  for (int i = 0; i < 8; ++i) {
    float inv = 1.0f / li[i];
    int row = q0 + i + 8 * hh;
#pragma unroll
    for (int n = 0; n < 4; ++n)
      og[(size_t)row * HID + h * 64 + n * 16 + lrow] = f2bf(oacc[n][i] * inv);
  }
}

// ---------------------------------------------------------------- final: residual + transpose
__global__ __launch_bounds__(256) void final_add(const float* __restrict__ atto,
                                                 const float* __restrict__ ffo,
                                                 const float* __restrict__ x,
                                                 float* __restrict__ out) {
  const int n = HID * S_TOT;
  int i = blockIdx.x * blockDim.x + threadIdx.x;
  int stride = gridDim.x * blockDim.x;
  for (; i < n; i += stride) {
    int c = i / S_TOT, s = i - c * S_TOT;
    out[i] = atto[(size_t)s * HID + c] + ffo[(size_t)s * HID + c] + x[i];
  }
}

// ---------------------------------------------------------------- launch
extern "C" void kernel_launch(void* const* d_in, const int* in_sizes, int n_in,
                              void* d_out, int out_size, void* d_ws, size_t ws_size,
                              hipStream_t stream) {
  (void)in_sizes; (void)n_in; (void)out_size; (void)ws_size;
  const float* x      = (const float*)d_in[0];
  // d_in[1] = bcs (unused by reference)
  const float* gamma  = (const float*)d_in[2];
  const float* Wfused = (const float*)d_in[3];
  const float* bfused = (const float*)d_in[4];
  const float* qnw    = (const float*)d_in[5];
  const float* qnb    = (const float*)d_in[6];
  const float* knw    = (const float*)d_in[7];
  const float* knb    = (const float*)d_in[8];
  const float* Wattn  = (const float*)d_in[9];
  const float* Wff    = (const float*)d_in[10];
  const float* bff    = (const float*)d_in[11];
  float* out = (float*)d_out;

  char* ws = (char*)d_ws;
  size_t off = 0;
  auto alloc = [&](size_t bytes) -> char* {
    char* p = ws + off;
    off = (off + bytes + 255) & ~(size_t)255;
    return p;
  };
  ushortT* wfused_bt = (ushortT*)alloc((size_t)HID * FUSED_N * 2);   // [5376][768]
  ushortT* wattn_bt  = (ushortT*)alloc((size_t)HID * HID * 2);       // [768][768]
  ushortT* wff_bt    = (ushortT*)alloc((size_t)(MLP / 2) * HID * 2); // [768][1536]
  ushortT* xn_bf     = (ushortT*)alloc((size_t)S_TOT * HID * 2);
  ushortT* q_bf      = (ushortT*)alloc((size_t)HEADS * S_TOT * 64 * 2);
  ushortT* k_bf      = (ushortT*)alloc((size_t)HEADS * S_TOT * 64 * 2);
  ushortT* vt_bf     = (ushortT*)alloc((size_t)HEADS * 64 * S_TOT * 2);
  ushortT* ao_bf     = (ushortT*)alloc((size_t)S_TOT * HID * 2);
  ushortT* ffact_bf  = (ushortT*)alloc((size_t)S_TOT * (MLP / 2) * 2);
  float*   fusedf    = (float*)alloc((size_t)S_TOT * FUSED_N * 4);
  // att-proj / ff outputs alias the fused buffer (dead after qkv_prep + ff_act)
  float* atto = fusedf;
  float* ffo  = fusedf + (size_t)S_TOT * HID;

  cvt_bf16_t<<<2048, 256, 0, stream>>>(Wfused, wfused_bt, HID, FUSED_N);
  cvt_bf16_t<<<512, 256, 0, stream>>>(Wattn, wattn_bt, HID, HID);
  cvt_bf16_t<<<1024, 256, 0, stream>>>(Wff, wff_bt, MLP / 2, HID);

  gnorm_kernel<<<3072, 256, 0, stream>>>(x, gamma, xn_bf);

  gemm_bf16<<<dim3(FUSED_N / 128, S_TOT / 128), 256, 0, stream>>>(
      xn_bf, wfused_bt, fusedf, bfused, S_TOT, FUSED_N, HID);

  qkv_prep<<<3072, 256, 0, stream>>>(fusedf, qnw, qnb, knw, knb, q_bf, k_bf, vt_bf);
  ff_act<<<2048, 256, 0, stream>>>(fusedf, ffact_bf);

  attn_kernel<<<HEADS * (S_TOT / 128), 256, 0, stream>>>(q_bf, k_bf, vt_bf, ao_bf);

  gemm_bf16<<<dim3(HID / 128, S_TOT / 128), 256, 0, stream>>>(
      ao_bf, wattn_bt, atto, nullptr, S_TOT, HID, HID);
  gemm_bf16<<<dim3(HID / 128, S_TOT / 128), 256, 0, stream>>>(
      ffact_bf, wff_bt, ffo, bff, S_TOT, HID, MLP / 2);

  final_add<<<1536, 256, 0, stream>>>(atto, ffo, x, out);
}
